// SeqToSeqAtten_16166256902631
// MI455X (gfx1250) — compile-verified, matrix operand order checked by reading, not run on hardware
//
#include <hip/hip_runtime.h>

// ---------------------------------------------------------------------------
// Flash-attention style fused kernel for
//   scores = einsum('cbh,qbh->bcq'); softmax(q); out = einsum('bcq,qbh->cbh')
// C_LEN=2048, Q_LEN=512, B=32, H=256, all fp32 in/out.
// Math: f16 WMMA (v_wmma_f32_16x16x32_f16) with f32 accumulation + f32 softmax.
// ---------------------------------------------------------------------------

typedef __attribute__((ext_vector_type(16))) _Float16 v16h;
typedef __attribute__((ext_vector_type(8)))  _Float16 v8h;
typedef __attribute__((ext_vector_type(8)))  float    v8f;

#define C_LEN   2048
#define Q_LEN   512
#define BATCH   32
#define HDIM    256
#define QCHUNK  64
#define NWAVES  8
#define LOG2E   1.44269504088896340736f

// LDS layout in halves:
//   [0,        16384) : Vrow  [64 q-rows][256 h]   (B operand for S = C*K^T)
//   [16384,    32768) : Vtr   [256 h]   [64 q]     (B operand for O += P*V)
//   [32768,    40960) : per-wave P staging [16][64]
#define V_ROW_OFF   0
#define V_TR_OFF    16384
#define P_OFF       32768
#define SMEM_HALVES 40960

extern __shared__ _Float16 smem[];

// Load a 16x32 f16 operand fragment (A or B layout) from row-major LDS.
// Per the CDNA5 ISA layout (CDNA5 ISA, 16-bit A 16x32): lane group g reads
// halves K = koff + g*8 + [0..8) and K = koff + 16 + g*8 + [0..8) from its row.
__device__ __forceinline__ v16h load_frag(const _Float16* row, int koff, int g) {
    v8h lo = *(const v8h*)(row + koff + g * 8);
    v8h hi = *(const v8h*)(row + koff + 16 + g * 8);
    return __builtin_shufflevector(lo, hi,
                                   0, 1, 2, 3, 4, 5, 6, 7,
                                   8, 9, 10, 11, 12, 13, 14, 15);
}

__global__ __launch_bounds__(256)
void seq2seq_attn_kernel(const float* __restrict__ content,
                         const float* __restrict__ question,
                         const int*   __restrict__ qmask,
                         float*       __restrict__ out)
{
    const int tid  = threadIdx.x;
    const int lane = tid & 31;
    const int wave = tid >> 5;
    const int g    = lane >> 4;   // half-wave group (selects K sub-range)
    const int ln   = lane & 15;   // row/col index within 16

    const int b  = blockIdx.y;                       // batch
    const int c0 = blockIdx.x * (16 * NWAVES) + wave * 16;  // this wave's 16 c-rows

    _Float16* Vrow = smem + V_ROW_OFF;
    _Float16* Vtr  = smem + V_TR_OFF;
    _Float16* Pst  = smem + P_OFF + wave * (16 * 64);

    // ---- Load this wave's content tile (16 x 256) once, f32 -> f16, A layout ----
    v16h afrag[8];
    {
        const float* crow = content + ((size_t)(c0 + ln) * BATCH + b) * HDIM;
#pragma unroll
        for (int kc = 0; kc < 8; ++kc) {
            const float* p0 = crow + kc * 32 + g * 8;
            const float* p1 = crow + kc * 32 + 16 + g * 8;
            v16h f;
#pragma unroll
            for (int i = 0; i < 8; ++i) f[i] = (_Float16)p0[i];
#pragma unroll
            for (int i = 0; i < 8; ++i) f[8 + i] = (_Float16)p1[i];
            afrag[kc] = f;
        }
    }

    // ---- Output accumulator (16 rows x 256 h) and online-softmax stats ----
    v8f o[16];
#pragma unroll
    for (int i = 0; i < 16; ++i) o[i] = (v8f){0.f, 0.f, 0.f, 0.f, 0.f, 0.f, 0.f, 0.f};
    float mmax[8], lsum[8];
#pragma unroll
    for (int v = 0; v < 8; ++v) { mmax[v] = -1e30f; lsum[v] = 0.f; }

    for (int jc = 0; jc < Q_LEN / QCHUNK; ++jc) {
        const int qbase = jc * QCHUNK;

        __syncthreads();  // previous chunk's consumers done before overwrite

        // ---- Cooperatively stage question chunk in LDS (row-major + transposed) ----
        {
            const int r  = tid >> 2;          // q-row within chunk: 0..63
            const int cc = (tid & 3) * 64;    // h-column start
            const float* src = question + ((size_t)(qbase + r) * BATCH + b) * HDIM + cc;
#pragma unroll 8
            for (int i = 0; i < 64; ++i) {
                _Float16 h = (_Float16)src[i];
                Vrow[r * 256 + cc + i] = h;
                Vtr[(cc + i) * 64 + r] = h;
            }
        }
        __syncthreads();

        // ---- Additive mask bias per score column ----
        float bias[4];
#pragma unroll
        for (int t = 0; t < 4; ++t)
            bias[t] = qmask[b * Q_LEN + qbase + t * 16 + ln] ? 0.f : -1e30f;

        // ---- S(16x64) = C_tile(16x256) @ K_chunk^T(256x64), f32 accum ----
        v8f s[4];
#pragma unroll
        for (int t = 0; t < 4; ++t) {
            v8f acc = (v8f){0.f, 0.f, 0.f, 0.f, 0.f, 0.f, 0.f, 0.f};
            const _Float16* brow = Vrow + (t * 16 + ln) * 256;
#pragma unroll
            for (int kc = 0; kc < 8; ++kc) {
                v16h bf = load_frag(brow, kc * 32, g);
                acc = __builtin_amdgcn_wmma_f32_16x16x32_f16(
                    false, afrag[kc], false, bf, (short)0, acc, false, false);
            }
#pragma unroll
            for (int i = 0; i < 8; ++i) acc[i] += bias[t];
            s[t] = acc;
        }

        // ---- Online softmax update (rows live per-VGPR, reduce over 16 lanes) ----
        float alpha[8];
#pragma unroll
        for (int v = 0; v < 8; ++v) {
            float cur = fmaxf(fmaxf(s[0][v], s[1][v]), fmaxf(s[2][v], s[3][v]));
#pragma unroll
            for (int off = 8; off >= 1; off >>= 1)
                cur = fmaxf(cur, __shfl_xor(cur, off, 32));
            const float mn = fmaxf(mmax[v], cur);
            alpha[v] = exp2f((mmax[v] - mn) * LOG2E);
            mmax[v] = mn;
            float rs = 0.f;
#pragma unroll
            for (int t = 0; t < 4; ++t) {
                float p = exp2f((s[t][v] - mn) * LOG2E);
                s[t][v] = p;
                rs += p;
            }
#pragma unroll
            for (int off = 8; off >= 1; off >>= 1)
                rs += __shfl_xor(rs, off, 32);
            lsum[v] = lsum[v] * alpha[v] + rs;
        }

        // ---- Re-shape P: C/D layout -> f16 A layout via per-wave LDS patch ----
#pragma unroll
        for (int t = 0; t < 4; ++t)
#pragma unroll
            for (int v = 0; v < 8; ++v)
                Pst[(v + g * 8) * 64 + t * 16 + ln] = (_Float16)s[t][v];

        // Rescale running output by alpha while the DS stores land
#pragma unroll
        for (int ht = 0; ht < 16; ++ht) {
            v8f tmp = o[ht];
#pragma unroll
            for (int v = 0; v < 8; ++v) tmp[v] *= alpha[v];
            o[ht] = tmp;
        }

        v16h pfrag0 = load_frag(Pst + ln * 64, 0, g);
        v16h pfrag1 = load_frag(Pst + ln * 64, 32, g);

        // ---- O(16x256) += P(16x64) @ V_chunk(64x256) ----
#pragma unroll
        for (int ht = 0; ht < 16; ++ht) {
            const _Float16* brow = Vtr + (ht * 16 + ln) * 64;
            v8f acc = o[ht];
            acc = __builtin_amdgcn_wmma_f32_16x16x32_f16(
                false, pfrag0, false, load_frag(brow, 0, g), (short)0, acc, false, false);
            acc = __builtin_amdgcn_wmma_f32_16x16x32_f16(
                false, pfrag1, false, load_frag(brow, 32, g), (short)0, acc, false, false);
            o[ht] = acc;
        }
    }

    // ---- Normalize and write out (coalesced: 16 consecutive h per half-wave) ----
#pragma unroll
    for (int v = 0; v < 8; ++v) {
        const float inv = 1.f / lsum[v];
        const int c = c0 + v + g * 8;
        float* dst = out + ((size_t)c * BATCH + b) * HDIM + ln;
#pragma unroll
        for (int ht = 0; ht < 16; ++ht)
            dst[ht * 16] = o[ht][v] * inv;
    }
}

extern "C" void kernel_launch(void* const* d_in, const int* in_sizes, int n_in,
                              void* d_out, int out_size, void* d_ws, size_t ws_size,
                              hipStream_t stream) {
    const float* content  = (const float*)d_in[0];  // (2048, 32, 256) f32
    const float* question = (const float*)d_in[1];  // (512, 32, 256) f32
    const int*   qmask    = (const int*)d_in[2];    // (32, 512) i32
    float*       out      = (float*)d_out;          // (2048, 32, 256) f32

    dim3 grid(C_LEN / (16 * NWAVES), BATCH);        // (16, 32) workgroups
    dim3 block(32 * NWAVES);                        // 256 threads = 8 waves
    size_t shmem = SMEM_HALVES * sizeof(_Float16);  // 80 KB dynamic LDS

    seq2seq_attn_kernel<<<grid, block, shmem, stream>>>(content, question, qmask, out);
}
